// backbone_11304353923566
// MI455X (gfx1250) — compile-verified, matrix operand order checked by reading, not run on hardware
//
#include <hip/hip_runtime.h>

// ---------------------------------------------------------------------------
// PointNet++ backbone for MI455X (gfx1250, wave32).
// All MLP layers: f16-packed activations/weights -> v_wmma_f32_16x16x32_f16
// with f32 accumulate. GEMM inner loop is branch-free: 6x global_load_b128 +
// 2x WMMA per 32-wide K chunk (16x32 output tile per wave).
// ---------------------------------------------------------------------------

typedef __attribute__((ext_vector_type(16))) _Float16 v16h;
typedef __attribute__((ext_vector_type(8)))  _Float16 v8h;
typedef __attribute__((ext_vector_type(8)))  float    v8f;

// CDNA5 16-bit A/B fragment: lane = (M or N) mod 16; lane-half h picks K groups
// [h*8 .. h*8+7] (VGPRs 0-3) and [16+h*8 .. 16+h*8+7] (VGPRs 4-7), i.e. per lane
// two contiguous 8-half runs -> two aligned b128 loads.
__device__ __forceinline__ v16h frag16(v8h lo, v8h hi) {
  return __builtin_shufflevector(lo, hi, 0, 1, 2, 3, 4, 5, 6, 7,
                                         8, 9, 10, 11, 12, 13, 14, 15);
}

// Y[m,n] = sum_k X[m,k]*Wp[n,k] + bias[n]
// X: f16, row stride Kp (mult of 32, zero-padded). Wp: f16, Np x Kp, zero-padded.
// Requires M % 128 == 0 (true for every call site). Block=256 (8 waves), each
// wave computes a 16x32 tile: grid = (M/128, Np/32).
__global__ void gemm_f16_kernel(const _Float16* __restrict__ X,
                                const _Float16* __restrict__ Wp,
                                const float* __restrict__ bias,
                                float* __restrict__ Y,
                                int M, int Kp, int Nout) {
  int wave = threadIdx.x >> 5;
  int lane = threadIdx.x & 31;
  int m0 = (blockIdx.x * 8 + wave) << 4;
  int n0 = blockIdx.y << 5;
  int h  = lane >> 4;
  int lm = lane & 15;
  const _Float16* xr  = X  + (size_t)(m0 + lm) * Kp + h * 8;
  const _Float16* wr0 = Wp + (size_t)(n0 + lm) * Kp + h * 8;
  const _Float16* wr1 = wr0 + (size_t)16 * Kp;
  v8f acc0 = {0.f, 0.f, 0.f, 0.f, 0.f, 0.f, 0.f, 0.f};
  v8f acc1 = {0.f, 0.f, 0.f, 0.f, 0.f, 0.f, 0.f, 0.f};
  for (int kb = 0; kb < Kp; kb += 32) {
    v8h alo  = *(const v8h*)(xr + kb);
    v8h ahi  = *(const v8h*)(xr + kb + 16);
    v8h b0lo = *(const v8h*)(wr0 + kb);
    v8h b0hi = *(const v8h*)(wr0 + kb + 16);
    v8h b1lo = *(const v8h*)(wr1 + kb);
    v8h b1hi = *(const v8h*)(wr1 + kb + 16);
    v16h a  = frag16(alo, ahi);
    v16h b0 = frag16(b0lo, b0hi);
    v16h b1 = frag16(b1lo, b1hi);
    acc0 = __builtin_amdgcn_wmma_f32_16x16x32_f16(false, a, false, b0, (short)0,
                                                  acc0, false, false);
    acc1 = __builtin_amdgcn_wmma_f32_16x16x32_f16(false, a, false, b1, (short)0,
                                                  acc1, false, false);
  }
  // C/D layout: VGPR j -> row m0 + j + h*8, col n0 + (lane&15)
  int nc0 = n0 + lm;
  int nc1 = n0 + 16 + lm;
  float bv0 = (nc0 < Nout) ? bias[nc0] : 0.f;
  float bv1 = (nc1 < Nout) ? bias[nc1] : 0.f;
#pragma unroll
  for (int j = 0; j < 8; ++j) {
    int mr = m0 + j + h * 8;
    if (nc0 < Nout) Y[(size_t)mr * Nout + nc0] = acc0[j] + bv0;
    if (nc1 < Nout) Y[(size_t)mr * Nout + nc1] = acc1[j] + bv1;
  }
}

// Pack f32 weights (Nout,K) -> f16 (Np,Kp), zero-padded in both dims.
__global__ void pack_w_kernel(const float* __restrict__ W, _Float16* __restrict__ Wp,
                              int Nout, int K, int Kp, long long total) {
  long long i = (long long)blockIdx.x * blockDim.x + threadIdx.x;
  if (i >= total) return;
  int n = (int)(i / Kp), k = (int)(i % Kp);
  Wp[i] = (n < Nout && k < K) ? (_Float16)W[(size_t)n * K + k] : (_Float16)0.f;
}

// ----------------- BatchNorm (training-mode stats over all M rows) ----------
__global__ void bn_stats_kernel(const float* __restrict__ Y, float* __restrict__ part,
                                int M, int C) {
  extern __shared__ float sm[];
  float* ssum = sm;
  float* ssq  = sm + C;
  for (int c = threadIdx.x; c < C; c += blockDim.x) { ssum[c] = 0.f; ssq[c] = 0.f; }
  __syncthreads();
  for (int r = blockIdx.x; r < M; r += gridDim.x)
    for (int c = threadIdx.x; c < C; c += blockDim.x) {
      float v = Y[(size_t)r * C + c];
      ssum[c] += v;                     // channel owned by exactly one thread
      ssq[c]  += v * v;
    }
  __syncthreads();
  for (int c = threadIdx.x; c < C; c += blockDim.x) {
    part[(size_t)blockIdx.x * 2 * C + c]     = ssum[c];
    part[(size_t)blockIdx.x * 2 * C + C + c] = ssq[c];
  }
}

__global__ void bn_finalize_kernel(const float* __restrict__ part,
                                   const float* __restrict__ g,
                                   const float* __restrict__ beta,
                                   float* __restrict__ ss, int M, int C, int npart) {
  for (int c = threadIdx.x; c < C; c += blockDim.x) {
    float s = 0.f, q = 0.f;
    for (int p = 0; p < npart; ++p) {       // fixed order -> deterministic
      s += part[(size_t)p * 2 * C + c];
      q += part[(size_t)p * 2 * C + C + c];
    }
    float mu  = s / (float)M;
    float var = q / (float)M - mu * mu;
    float sc  = g[c] * rsqrtf(var + 1e-5f);
    ss[c]     = sc;
    ss[C + c] = beta[c] - sc * mu;
  }
}

// Apply scale/shift + ReLU: overwrite f32 Y in place, optionally emit f16 copy
// (the next layer's activation; C is a multiple of 32 so no extra padding).
__global__ void bn_relu_kernel(float* __restrict__ Y, _Float16* __restrict__ Xn,
                               const float* __restrict__ ss, long long total, int C) {
  long long i = (long long)blockIdx.x * blockDim.x + threadIdx.x;
  if (i >= total) return;
  int c = (int)(i % C);
  float v = Y[i] * ss[c] + ss[C + c];
  v = v > 0.f ? v : 0.f;
  Y[i] = v;
  if (Xn) Xn[i] = (_Float16)v;
}

// ----------------- Farthest point sampling (one block per batch) ------------
__global__ void fps_kernel(const float* __restrict__ xyz, int* __restrict__ out_idx,
                           int N, int npoint) {
  int b = blockIdx.x;
  const float* p = xyz + (size_t)b * N * 3;
  extern __shared__ float sdist[];   // N floats
  __shared__ float rmax[256];
  __shared__ int   rarg[256];
  __shared__ int   far;
  for (int i = threadIdx.x; i < N; i += blockDim.x) sdist[i] = 1e10f;
  if (threadIdx.x == 0) far = 0;
  __syncthreads();
  for (int it = 0; it < npoint; ++it) {
    if (threadIdx.x == 0) out_idx[b * npoint + it] = far;
    __syncthreads();
    float cx = p[far * 3], cy = p[far * 3 + 1], cz = p[far * 3 + 2];
    float best = -1.f; int bi = 0x7fffffff;
    for (int i = threadIdx.x; i < N; i += blockDim.x) {
      float dx = p[i * 3] - cx, dy = p[i * 3 + 1] - cy, dz = p[i * 3 + 2] - cz;
      float d  = dx * dx + dy * dy + dz * dz;
      float dd = fminf(sdist[i], d);
      sdist[i] = dd;
      if (dd > best) { best = dd; bi = i; }
    }
    rmax[threadIdx.x] = best; rarg[threadIdx.x] = bi;
    __syncthreads();
    for (int s = blockDim.x >> 1; s > 0; s >>= 1) {
      if (threadIdx.x < s) {
        float ov = rmax[threadIdx.x + s]; int oi = rarg[threadIdx.x + s];
        if (ov > rmax[threadIdx.x] ||
            (ov == rmax[threadIdx.x] && oi < rarg[threadIdx.x])) {
          rmax[threadIdx.x] = ov; rarg[threadIdx.x] = oi;
        }
      }
      __syncthreads();
    }
    if (threadIdx.x == 0) far = rarg[0];
    __syncthreads();
  }
}

__global__ void gather_xyz_kernel(const float* __restrict__ xyz, const int* __restrict__ idx,
                                  float* __restrict__ out, int Bn, int Np, int N) {
  int t = blockIdx.x * blockDim.x + threadIdx.x;
  if (t >= Bn * Np) return;
  int b = t / Np;
  int j = idx[t];
  const float* p = xyz + ((size_t)b * N + j) * 3;
  out[t * 3 + 0] = p[0]; out[t * 3 + 1] = p[1]; out[t * 3 + 2] = p[2];
}

__global__ void ball_query_kernel(const float* __restrict__ xyz, const float* __restrict__ ctr,
                                  int* __restrict__ idx, int Bn, int S, int N,
                                  float r2, int nsample) {
  int t = blockIdx.x * blockDim.x + threadIdx.x;
  if (t >= Bn * S) return;
  int b = t / S;
  const float* c = ctr + (size_t)t * 3;
  const float* p = xyz + (size_t)b * N * 3;
  int* row = idx + (size_t)t * nsample;
  int cnt = 0, first = 0;
  for (int i = 0; i < N && cnt < nsample; ++i) {
    float dx = c[0] - p[i * 3], dy = c[1] - p[i * 3 + 1], dz = c[2] - p[i * 3 + 2];
    if (dx * dx + dy * dy + dz * dz <= r2) {
      if (cnt == 0) first = i;
      row[cnt++] = i;
    }
  }
  for (; cnt < nsample; ++cnt) row[cnt] = first;
}

// grouped[m] = [xyz[idx]-center (3), feats[idx] (Cf)] -> f16, zero-pad to Kp
__global__ void group_f16_kernel(const float* __restrict__ xyz, const float* __restrict__ feats,
                                 const float* __restrict__ ctr, const int* __restrict__ idx,
                                 _Float16* __restrict__ X, int Bn, int S, int K, int N,
                                 int Cf, int Kp) {
  int t = blockIdx.x * blockDim.x + threadIdx.x;
  if (t >= Bn * S * K) return;
  int bs = t / K;
  int b  = bs / S;
  int j  = idx[t];
  const float* p  = xyz + ((size_t)b * N + j) * 3;
  const float* c0 = ctr + (size_t)bs * 3;
  _Float16* row = X + (size_t)t * Kp;
  row[0] = (_Float16)(p[0] - c0[0]);
  row[1] = (_Float16)(p[1] - c0[1]);
  row[2] = (_Float16)(p[2] - c0[2]);
  for (int c = 0; c < Cf; ++c) row[3 + c] = (_Float16)feats[((size_t)b * N + j) * Cf + c];
  for (int c = 3 + Cf; c < Kp; ++c) row[c] = (_Float16)0.f;
}

// SA3 group_all: row = [xyz (3), feats (Cf)] (no centering), f16 + pad
__global__ void concat_all_f16_kernel(const float* __restrict__ xyz, const float* __restrict__ f,
                                      _Float16* __restrict__ X, int BS, int Cf, int Kp) {
  int t = blockIdx.x * blockDim.x + threadIdx.x;
  if (t >= BS) return;
  _Float16* row = X + (size_t)t * Kp;
  row[0] = (_Float16)xyz[t * 3];
  row[1] = (_Float16)xyz[t * 3 + 1];
  row[2] = (_Float16)xyz[t * 3 + 2];
  for (int c = 0; c < Cf; ++c) row[3 + c] = (_Float16)f[(size_t)t * Cf + c];
  for (int c = 3 + Cf; c < Kp; ++c) row[c] = (_Float16)0.f;
}

// FP with S==1: row = [f1[b,s] (C1), broadcast f2[b] (C2)], f16 (C1+C2 mult 32)
__global__ void bcast_concat_f16_kernel(const float* __restrict__ f1,
                                        const float* __restrict__ f2,
                                        _Float16* __restrict__ X, int Bn, int S,
                                        int C1, int C2) {
  int t = blockIdx.x * blockDim.x + threadIdx.x;
  if (t >= Bn * S) return;
  int b = t / S;
  _Float16* row = X + (size_t)t * (C1 + C2);
  for (int c = 0; c < C1; ++c) row[c]      = (_Float16)f1[(size_t)t * C1 + c];
  for (int c = 0; c < C2; ++c) row[C1 + c] = (_Float16)f2[(size_t)b * C2 + c];
}

// FP 3-NN inverse-distance interpolation; row = [f1 (C1), interp(f2) (C2)], f16
__global__ void interp_f16_kernel(const float* __restrict__ xyz1, const float* __restrict__ xyz2,
                                  const float* __restrict__ f1, const float* __restrict__ f2,
                                  _Float16* __restrict__ X, int Bn, int N1, int S,
                                  int C1, int C2) {
  int t = blockIdx.x * blockDim.x + threadIdx.x;
  if (t >= Bn * N1) return;
  int b = t / N1;
  const float* p = xyz1 + (size_t)t * 3;
  float d0 = 1e30f, d1 = 1e30f, d2 = 1e30f;
  int   i0 = 0,     i1 = 0,     i2 = 0;
  for (int s = 0; s < S; ++s) {
    const float* q = xyz2 + ((size_t)b * S + s) * 3;
    float dx = p[0] - q[0], dy = p[1] - q[1], dz = p[2] - q[2];
    float dd = dx * dx + dy * dy + dz * dz;
    if (dd < d0)      { d2 = d1; i2 = i1; d1 = d0; i1 = i0; d0 = dd; i0 = s; }
    else if (dd < d1) { d2 = d1; i2 = i1; d1 = dd; i1 = s; }
    else if (dd < d2) { d2 = dd; i2 = s; }
  }
  float w0 = 1.f / (d0 + 1e-8f), w1 = 1.f / (d1 + 1e-8f), w2 = 1.f / (d2 + 1e-8f);
  float ws = w0 + w1 + w2;
  w0 /= ws; w1 /= ws; w2 /= ws;
  _Float16* row = X + (size_t)t * (C1 + C2);
  for (int c = 0; c < C1; ++c) row[c] = (_Float16)f1[(size_t)t * C1 + c];
  const float* r0 = f2 + ((size_t)b * S + i0) * C2;
  const float* r1 = f2 + ((size_t)b * S + i1) * C2;
  const float* r2 = f2 + ((size_t)b * S + i2) * C2;
  for (int c = 0; c < C2; ++c)
    row[C1 + c] = (_Float16)(w0 * r0[c] + w1 * r1[c] + w2 * r2[c]);
}

__global__ void maxpool_kernel(const float* __restrict__ Y, float* __restrict__ out,
                               int BS, int K, int C) {
  long long i = (long long)blockIdx.x * blockDim.x + threadIdx.x;
  if (i >= (long long)BS * C) return;
  int bs = (int)(i / C), c = (int)(i % C);
  float m = -3.4e38f;
  for (int k = 0; k < K; ++k) m = fmaxf(m, Y[((size_t)bs * K + k) * C + c]);
  out[i] = m;
}

// ---------------------------------------------------------------------------
extern "C" void kernel_launch(void* const* d_in, const int* in_sizes, int n_in,
                              void* d_out, int out_size, void* d_ws, size_t ws_size,
                              hipStream_t stream) {
  (void)in_sizes; (void)n_in; (void)out_size; (void)ws_size;
  const int B = 16, N = 2048;
  const int NSTAT = 128;
  const float* x = (const float*)d_in[0];

  size_t off = 0;
  auto carve = [&](size_t nbytes) -> void* {
    void* p = (char*)d_ws + off;
    off += (nbytes + 255) & ~(size_t)255;
    return p;
  };
  const size_t FMAX = (size_t)524288 * 128;             // max f16 activation elems
  _Float16* fA   = (_Float16*)carve(FMAX * 2);
  _Float16* fB   = (_Float16*)carve(FMAX * 2);
  float*    bufY = (float*)   carve(FMAX * 4);          // f32 GEMM output scratch
  _Float16* wbuf = (_Float16*)carve((size_t)4 << 20);   // packed weights (~2.8MB)
  int*   fps1 = (int*)  carve((size_t)B * 512 * 4);
  float* l1x  = (float*)carve((size_t)B * 512 * 3 * 4);
  int*   idx1 = (int*)  carve((size_t)B * 512 * 64 * 4);
  float* l1f  = (float*)carve((size_t)B * 512 * 128 * 4);
  int*   fps2 = (int*)  carve((size_t)B * 128 * 4);
  float* l2x  = (float*)carve((size_t)B * 128 * 3 * 4);
  int*   idx2 = (int*)  carve((size_t)B * 128 * 64 * 4);
  float* l2f  = (float*)carve((size_t)B * 128 * 256 * 4);
  float* l3f  = (float*)carve((size_t)B * 1024 * 4);
  float* l4f  = (float*)carve((size_t)B * 128 * 256 * 4);
  float* l5f  = (float*)carve((size_t)B * 512 * 128 * 4);
  float* part = (float*)carve((size_t)NSTAT * 2 * 1024 * 4);
  float* ss   = (float*)carve((size_t)2 * 1024 * 4);

  auto cdiv  = [](long long a, long long b) { return (int)((a + b - 1) / b); };
  auto pad32 = [](int v) { return (v + 31) & ~31; };
  size_t woff = 0;

  // One MLP layer: pack W -> WMMA GEMM+bias -> BN stats -> BN+ReLU apply.
  // Yf32 receives the f32 post-BN result; Xout (may be null) the f16 copy.
  auto mlp_layer = [&](const _Float16* Xin, _Float16* Xout, float* Yf32,
                       int pbase, int M, int K, int Nc) {
    const float* W  = (const float*)d_in[pbase];
    const float* bi = (const float*)d_in[pbase + 1];
    const float* g  = (const float*)d_in[pbase + 2];
    const float* be = (const float*)d_in[pbase + 3];
    int Kp = pad32(K), Np = pad32(Nc);
    _Float16* Wp = wbuf + woff;
    woff += (size_t)Np * Kp;
    long long wtotal = (long long)Np * Kp;
    pack_w_kernel<<<cdiv(wtotal, 256), 256, 0, stream>>>(W, Wp, Nc, K, Kp, wtotal);
    dim3 grid(M / 128, Np / 32);
    gemm_f16_kernel<<<grid, 256, 0, stream>>>(Xin, Wp, bi, Yf32, M, Kp, Nc);
    bn_stats_kernel<<<NSTAT, 256, 2 * Nc * sizeof(float), stream>>>(Yf32, part, M, Nc);
    bn_finalize_kernel<<<1, 256, 0, stream>>>(part, g, be, ss, M, Nc, NSTAT);
    long long total = (long long)M * Nc;
    bn_relu_kernel<<<cdiv(total, 256), 256, 0, stream>>>(Yf32, Xout, ss, total, Nc);
  };

  // ---------------- SA1: npoint=512, r=0.2, nsample=64, [3,64,64,128] -------
  fps_kernel<<<B, 256, N * sizeof(float), stream>>>(x, fps1, N, 512);
  gather_xyz_kernel<<<cdiv(B * 512, 256), 256, 0, stream>>>(x, fps1, l1x, B, 512, N);
  ball_query_kernel<<<cdiv(B * 512, 256), 256, 0, stream>>>(x, l1x, idx1, B, 512, N,
                                                            0.04f, 64);
  int M1 = B * 512 * 64;
  group_f16_kernel<<<cdiv(M1, 256), 256, 0, stream>>>(x, nullptr, l1x, idx1, fA,
                                                      B, 512, 64, N, 0, 32);
  mlp_layer(fA, fB, bufY, 1, M1, 3, 64);
  mlp_layer(fB, fA, bufY, 5, M1, 64, 64);
  mlp_layer(fA, fB, bufY, 9, M1, 64, 128);
  maxpool_kernel<<<cdiv((long long)B * 512 * 128, 256), 256, 0, stream>>>(
      bufY, l1f, B * 512, 64, 128);

  // ---------------- SA2: npoint=128, r=0.4, nsample=64, [131,128,128,256] ---
  fps_kernel<<<B, 256, 512 * sizeof(float), stream>>>(l1x, fps2, 512, 128);
  gather_xyz_kernel<<<cdiv(B * 128, 256), 256, 0, stream>>>(l1x, fps2, l2x, B, 128, 512);
  ball_query_kernel<<<cdiv(B * 128, 256), 256, 0, stream>>>(l1x, l2x, idx2, B, 128, 512,
                                                            0.16f, 64);
  int M2 = B * 128 * 64;
  group_f16_kernel<<<cdiv(M2, 256), 256, 0, stream>>>(l1x, l1f, l2x, idx2, fA,
                                                      B, 128, 64, 512, 128, 160);
  mlp_layer(fA, fB, bufY, 13, M2, 131, 128);
  mlp_layer(fB, fA, bufY, 17, M2, 128, 128);
  mlp_layer(fA, fB, bufY, 21, M2, 128, 256);
  maxpool_kernel<<<cdiv((long long)B * 128 * 256, 256), 256, 0, stream>>>(
      bufY, l2f, B * 128, 64, 256);

  // ---------------- SA3: group_all, [259,256,512,1024] ----------------------
  int M3 = B * 128;
  concat_all_f16_kernel<<<cdiv(M3, 256), 256, 0, stream>>>(l2x, l2f, fA, M3, 256, 288);
  mlp_layer(fA, fB, bufY, 25, M3, 259, 256);
  mlp_layer(fB, fA, bufY, 29, M3, 256, 512);
  mlp_layer(fA, fB, bufY, 33, M3, 512, 1024);
  maxpool_kernel<<<cdiv((long long)B * 1024, 256), 256, 0, stream>>>(
      bufY, l3f, B, 128, 1024);

  // ---------------- FP3: S==1 broadcast, [1280,256,256] ---------------------
  bcast_concat_f16_kernel<<<cdiv(M3, 256), 256, 0, stream>>>(l2f, l3f, fA, B, 128,
                                                             256, 1024);
  mlp_layer(fA, fB, bufY, 37, M3, 1280, 256);
  mlp_layer(fB, nullptr, l4f, 41, M3, 256, 256);

  // ---------------- FP2: interp l2->l1, [384,256,128] -----------------------
  int M4 = B * 512;
  interp_f16_kernel<<<cdiv(M4, 256), 256, 0, stream>>>(l1x, l2x, l1f, l4f, fA,
                                                       B, 512, 128, 128, 256);
  mlp_layer(fA, fB, bufY, 45, M4, 384, 256);
  mlp_layer(fB, nullptr, l5f, 49, M4, 256, 128);

  // ---------------- FP1: interp l1->x, feats1=None, [128,128,128,128] -------
  int M5 = B * N;
  interp_f16_kernel<<<cdiv(M5, 256), 256, 0, stream>>>(x, l1x, nullptr, l5f, fA,
                                                       B, N, 512, 0, 128);
  mlp_layer(fA, fB, bufY, 53, M5, 128, 128);
  mlp_layer(fB, fA, bufY, 57, M5, 128, 128);
  mlp_layer(fA, fB, bufY, 61, M5, 128, 128);

  // ---------------- fc1 (BN+ReLU) + head (plain linear) ---------------------
  mlp_layer(fB, fA, bufY, 65, M5, 128, 128);
  {
    const float* Wh = (const float*)d_in[69];
    const float* bh = (const float*)d_in[70];
    int Kp = 128, Np = 32;
    _Float16* Wp = wbuf + woff;
    woff += (size_t)Np * Kp;
    long long wtotal = (long long)Np * Kp;
    pack_w_kernel<<<cdiv(wtotal, 256), 256, 0, stream>>>(Wh, Wp, 3, 128, Kp, wtotal);
    dim3 grid(M5 / 128, Np / 32);
    gemm_f16_kernel<<<grid, 256, 0, stream>>>(fA, Wp, bh, (float*)d_out, M5, Kp, 3);
  }
}
